// MultiCrossAlign_head_atttrans_res1sepalign_17755394801962
// MI455X (gfx1250) — matrix operand order repaired and checked by reading, not compiled, not run on hardware
//
#include <hip/hip_runtime.h>

typedef __attribute__((ext_vector_type(16))) _Float16 v16h;
typedef __attribute__((ext_vector_type(8)))  _Float16 v8h;
typedef __attribute__((ext_vector_type(8)))  float    v8f;

#define LREL(x) ((x) > 0.0f ? (x) : 0.1f * (x))

__device__ __forceinline__ v8f wmma_f16(v16h a, v16h b, v8f c) {
  return __builtin_amdgcn_wmma_f32_16x16x32_f16(false, a, false, b, (short)0, c, false, false);
}

// A fragment: 16x32 f16, src row-major with leading dim ld (halves)
__device__ __forceinline__ v16h frag_a(const _Float16* __restrict__ src, int ld) {
  int l = threadIdx.x & 31, m = l & 15, g = l >> 4;
  v16h a;
#pragma unroll
  for (int h = 0; h < 16; ++h) {
    int k = h + 16 * g;
    a[h] = src[m * ld + k];
  }
  return a;
}

// B fragment from pre-packed weights: tile = [lane][16 halves] contiguous
__device__ __forceinline__ v16h frag_b_packed(const _Float16* __restrict__ tile) {
  return *(const v16h*)(tile + (threadIdx.x & 31) * 16);
}

// B fragment = transpose of row-major src (N rows x K=32 cols): contiguous in k
__device__ __forceinline__ v16h frag_bt(const _Float16* __restrict__ src, int ld) {
  int l = threadIdx.x & 31, n = l & 15, g = l >> 4;
  v16h b;
#pragma unroll
  for (int h = 0; h < 16; ++h) b[h] = src[n * ld + 16 * g + h];
  return b;
}

// A fragment with only K=16 valid (head dim), upper half zero-padded
__device__ __forceinline__ v16h frag_a_k16(const _Float16* __restrict__ src, int ld) {
  int l = threadIdx.x & 31, m = l & 15, g = l >> 4;
  v16h a;
#pragma unroll
  for (int h = 0; h < 16; ++h) {
    int k = h + 16 * g;
    a[h] = (k < 16) ? src[m * ld + k] : (_Float16)0.0f;
  }
  return a;
}

// B = transpose of row-major src (N rows x K cols), K=16 padded to 32
__device__ __forceinline__ v16h frag_bt_k16(const _Float16* __restrict__ src, int ld) {
  int l = threadIdx.x & 31, n = l & 15, g = l >> 4;
  v16h b;
#pragma unroll
  for (int h = 0; h < 16; ++h) {
    int k = h + 16 * g;
    b[h] = (k < 16) ? src[n * ld + k] : (_Float16)0.0f;
  }
  return b;
}

// ---------------------------------------------------------------------------
// Windowed cross-attention: one block (8 wave32) per 8x8 window.
// Packed weight tiles: [kc(2)][ni(4)][lane(32)][h(16)] per 64x64 matrix.
// ---------------------------------------------------------------------------
__global__ __launch_bounds__(256) void win_attn_kernel(
    const float* __restrict__ x, const float* __restrict__ y,
    const _Float16* __restrict__ Wq, const _Float16* __restrict__ Wkm,
    const _Float16* __restrict__ Wv, const _Float16* __restrict__ Wp,
    const float* __restrict__ qb, const float* __restrict__ kb,
    const float* __restrict__ vb, const float* __restrict__ pb,
    const float* __restrict__ btab,   // [225][4]
    float* __restrict__ feat, float* __restrict__ attnMean,
    int H, int W) {
  __shared__ _Float16 sA[64 * 64];   // masked X (f16) -> later O
  __shared__ _Float16 sB[64 * 64];   // Y (f16)        -> later softmax(S) f16
  __shared__ _Float16 sQ[64 * 64];   // [token][c]
  __shared__ _Float16 sK[64 * 64];   // [token][c]
  __shared__ _Float16 sVT[64 * 64];  // transposed: [c][token]
  __shared__ float    sS[64 * 64];   // scores f32 -> later projection f32
  __shared__ float    sMw[64];

  const int tid = threadIdx.x;
  const int wave = tid >> 5;
  const int nwx = W >> 3, nwy = H >> 3;
  const int wid = blockIdx.x;
  const int b = wid / (nwy * nwx);
  const int rr0 = wid % (nwy * nwx);
  const int wy = rr0 / nwx, wx = rr0 % nwx;
  const int h0 = wy << 3, w0 = wx << 3;

  // load X (f32 -> sS) and Y (f16 -> sB), token-major [t][c]
  for (int e = tid; e < 4096; e += 256) {
    int c = e >> 6, t = e & 63;
    size_t gidx = ((size_t)(b * 64 + c) * H + (h0 + (t >> 3))) * W + (w0 + (t & 7));
    sS[t * 64 + c] = x[gidx];
    sB[t * 64 + c] = (_Float16)y[gidx];
  }
  __syncthreads();
  if (tid < 64) {
    float s = 0.0f;
    for (int c = 0; c < 64; ++c) s += (sS[tid * 64 + c] > 0.95f) ? 0.0f : 1.0f;
    sMw[tid] = s * (1.0f / 64.0f);
  }
  __syncthreads();
  for (int e = tid; e < 4096; e += 256) {
    int c = e >> 6, t = e & 63;
    sA[t * 64 + c] = (_Float16)(sS[t * 64 + c] * sMw[t]);
  }
  __syncthreads();

  // Q = maskedX @ Wq + qb (scaled 1/sqrt(16)); K,V from Y; V stored transposed
  {
    const int l = tid & 31, n = l & 15, g = l >> 4;
#pragma unroll
    for (int t2 = 0; t2 < 2; ++t2) {
      int tile = wave * 2 + t2;
      int mi = tile >> 2, ni = tile & 3;
      v8f aq = {}, ak = {}, av = {};
#pragma unroll
      for (int kc = 0; kc < 2; ++kc) {
        v16h fx = frag_a(sA + mi * 16 * 64 + kc * 32, 64);
        v16h fy = frag_a(sB + mi * 16 * 64 + kc * 32, 64);
        int toff = (kc * 4 + ni) * 512;
        aq = wmma_f16(fx, frag_b_packed(Wq + toff), aq);
        ak = wmma_f16(fy, frag_b_packed(Wkm + toff), ak);
        av = wmma_f16(fy, frag_b_packed(Wv + toff), av);
      }
      float bq = qb[ni * 16 + n], bk = kb[ni * 16 + n], bv = vb[ni * 16 + n];
      v8h pv;
#pragma unroll
      for (int r = 0; r < 8; ++r) {
        int row = mi * 16 + r + 8 * g;
        sQ[row * 64 + ni * 16 + n] = (_Float16)((aq[r] + bq) * 0.25f);
        sK[row * 64 + ni * 16 + n] = (_Float16)(ak[r] + bk);
        pv[r] = (_Float16)(av[r] + bv);
      }
      // transposed V store: contiguous 8 halves (16B aligned)
      *(v8h*)(sVT + (ni * 16 + n) * 64 + mi * 16 + 8 * g) = pv;
    }
  }
  __syncthreads();

  for (int hh = 0; hh < 4; ++hh) {
    // scores S = q_h @ k_h^T, masked + relative bias
    {
      const int l = tid & 31, n = l & 15, g = l >> 4;
#pragma unroll
      for (int t2 = 0; t2 < 2; ++t2) {
        int tile = wave * 2 + t2;
        int mi = tile >> 2, ni = tile & 3;
        v8f acc = {};
        v16h fa = frag_a_k16(sQ + mi * 16 * 64 + hh * 16, 64);
        v16h fb = frag_bt_k16(sK + ni * 16 * 64 + hh * 16, 64);
        acc = wmma_f16(fa, fb, acc);
        int tk = ni * 16 + n;
        int i2 = tk >> 3, j2 = tk & 7;
        float mk = sMw[tk];
#pragma unroll
        for (int r = 0; r < 8; ++r) {
          int tq = mi * 16 + r + 8 * g;
          int i1 = tq >> 3, j1 = tq & 7;
          int bidx = ((i1 - i2 + 7) * 15 + (j1 - j2 + 7)) * 4 + hh;
          sS[tq * 64 + tk] = acc[r] * (sMw[tq] * mk) + btab[bidx];
        }
      }
    }
    __syncthreads();
    // softmax per row; emit f16 probs + attention-mean accumulation
    if (tid < 64) {
      float* row = sS + tid * 64;
      float mx = -3.0e38f;
      for (int m2 = 0; m2 < 64; ++m2) mx = fmaxf(mx, row[m2]);
      float sum = 0.0f;
      for (int m2 = 0; m2 < 64; ++m2) { float ev = __expf(row[m2] - mx); row[m2] = ev; sum += ev; }
      float inv = 1.0f / sum;
      float* am = attnMean ? (attnMean + (size_t)wid * 4096 + tid * 64) : nullptr;
      for (int m2 = 0; m2 < 64; ++m2) {
        float p = row[m2] * inv;
        sB[tid * 64 + m2] = (_Float16)p;
        if (am) { if (hh == 0) am[m2] = 0.25f * p; else am[m2] += 0.25f * p; }
      }
    }
    __syncthreads();
    // out_h = S @ V_h -> columns [hh*16, hh*16+16) of sA
    if (wave < 4) {
      const int l = tid & 31, n = l & 15, g = l >> 4;
      int mi = wave;
      v8f acc = {};
#pragma unroll
      for (int kk = 0; kk < 64; kk += 32) {
        v16h fa = frag_a(sB + mi * 16 * 64 + kk, 64);
        v16h fb = frag_bt(sVT + hh * 16 * 64 + kk, 64);
        acc = wmma_f16(fa, fb, acc);
      }
#pragma unroll
      for (int r = 0; r < 8; ++r)
        sA[(mi * 16 + r + 8 * g) * 64 + hh * 16 + n] = (_Float16)acc[r];
    }
    __syncthreads();
  }

  // projection: P = O @ Wp + pb -> sS (f32)
  {
    const int l = tid & 31, n = l & 15, g = l >> 4;
#pragma unroll
    for (int t2 = 0; t2 < 2; ++t2) {
      int tile = wave * 2 + t2;
      int mi = tile >> 2, ni = tile & 3;
      v8f acc = {};
#pragma unroll
      for (int kc = 0; kc < 2; ++kc) {
        v16h fa = frag_a(sA + mi * 16 * 64 + kc * 32, 64);
        v16h fb = frag_b_packed(Wp + (kc * 4 + ni) * 512);
        acc = wmma_f16(fa, fb, acc);
      }
      float bp = pb[ni * 16 + n];
#pragma unroll
      for (int r = 0; r < 8; ++r)
        sS[(mi * 16 + r + 8 * g) * 64 + ni * 16 + n] = acc[r] + bp;
    }
  }
  __syncthreads();
  for (int e = tid; e < 4096; e += 256) {
    int c = e >> 6, t = e & 63;
    size_t gidx = ((size_t)(b * 64 + c) * H + (h0 + (t >> 3))) * W + (w0 + (t & 7));
    feat[gidx] = sS[t * 64 + c];
  }
}

// ---------------------------------------------------------------------------
// Attention transfer: out = mean_head(att) @ rearranged 16x16 tile of ta
// One block per 16x16 window; GEMM 64x256x64 (64 WMMA tiles).
// ---------------------------------------------------------------------------
__global__ __launch_bounds__(256) void atttrans_kernel(
    const float* __restrict__ ta, const float* __restrict__ attnMean,
    float* __restrict__ outp, int H, int W) {
  __shared__ _Float16 Ah[64 * 64];    // [n token][m token]
  __shared__ _Float16 BhT[256 * 64];  // transposed: [f][m token]
  const int tid = threadIdx.x;
  const int wave = tid >> 5;
  const int nwx = W >> 4, nwy = H >> 4;
  const int wid = blockIdx.x;
  const int b = wid / (nwy * nwx);
  const int rr0 = wid % (nwy * nwx);
  const int wy = rr0 / nwx, wx = rr0 % nwx;
  const int h0 = wy << 4, w0 = wx << 4;

  for (int e = tid; e < 4096; e += 256)
    Ah[e] = (_Float16)attnMean[(size_t)wid * 4096 + e];
  for (int e = tid; e < 16384; e += 256) {
    int c = e >> 8, rest = e & 255, ph = rest >> 4, pw2 = rest & 15;
    float v = ta[((size_t)(b * 64 + c) * H + (h0 + ph)) * W + (w0 + pw2)];
    int m = ((ph >> 1) << 3) + (pw2 >> 1);
    int f = (((ph & 1) << 1) + (pw2 & 1)) * 64 + c;
    BhT[f * 64 + m] = (_Float16)v;
  }
  __syncthreads();

  const int l = tid & 31, n = l & 15, g = l >> 4;
#pragma unroll
  for (int t2 = 0; t2 < 8; ++t2) {
    int tile = wave * 8 + t2;
    int mi = tile >> 4, ni = tile & 15;
    v8f acc = {};
#pragma unroll
    for (int kk = 0; kk < 64; kk += 32) {
      v16h fa = frag_a(Ah + mi * 16 * 64 + kk, 64);
      v16h fb = frag_bt(BhT + (ni * 16) * 64 + kk, 64);
      acc = wmma_f16(fa, fb, acc);
    }
    int f = ni * 16 + n;
    int c = f & 63, pq = f >> 6, p = pq >> 1, q = pq & 1;
#pragma unroll
    for (int r = 0; r < 8; ++r) {
      int m = mi * 16 + r + 8 * g;
      int a2 = m >> 3, b2 = m & 7;
      int gh = h0 + a2 * 2 + p, gw = w0 + b2 * 2 + q;
      outp[((size_t)(b * 64 + c) * H + gh) * W + gw] = acc[r];
    }
  }
}

// ---------------------------------------------------------------------------
// 3x3 conv (192->64) as implicit GEMM. Block computes 4 rows x 16 cols x 64oc.
// k-order: k = (ky*3+kx)*192 + ci  (tap-major; 32-chunks never cross a tap).
// Patch LDS layout pixel-major: P2[py][px][ci] so A fragments are contiguous.
// Weights pre-packed fragment-ready: [kc(54)][ni(4)][lane(32)][h(16)].
// ---------------------------------------------------------------------------
__global__ __launch_bounds__(256) void conv3_kernel(
    const float* __restrict__ in0, const float* __restrict__ in1,
    const float* __restrict__ in2, const _Float16* __restrict__ Wk,
    const float* __restrict__ bias, float* __restrict__ outp,
    int H, int W, int doLrelu) {
  __shared__ _Float16 P2[6 * 18 * 192];  // [py][px][ci]
  __shared__ float Of[64 * 64];          // [pixel][oc]
  const int tid = threadIdx.x;
  const int wave = tid >> 5;
  const int nbx = W >> 4, nby = H >> 2;
  const int bid = blockIdx.x;
  const int b = bid / (nby * nbx);
  const int rr0 = bid % (nby * nbx);
  const int by = rr0 / nbx, bx = rr0 % nbx;
  const int h0 = by << 2, w0 = bx << 4;

  for (int e = tid; e < 192 * 108; e += 256) {
    int ci = e / 108, rem = e % 108, py = rem / 18, px = rem % 18;
    int gh = h0 + py - 1, gw = w0 + px - 1;
    float v = 0.0f;
    if (gh >= 0 && gh < H && gw >= 0 && gw < W) {
      const float* src = (ci < 64) ? in0 : (ci < 128) ? in1 : in2;
      int cc = ci & 63;
      v = src[((size_t)(b * 64 + cc) * H + gh) * W + gw];
    }
    P2[(py * 18 + px) * 192 + ci] = (_Float16)v;
  }
  __syncthreads();

  const int l = tid & 31, lm = l & 15, g = l >> 4;
  const int mi = wave >> 1;          // pixel-row tile (4 rows)
  const int nt0 = (wave & 1) * 2;    // first of two 16-wide oc tiles
  v8f acc0 = {}, acc1 = {};
  int kc = 0;
#pragma unroll
  for (int tap = 0; tap < 9; ++tap) {
    const int ky = tap / 3, kx = tap % 3;
    const _Float16* prow = P2 + ((mi + ky) * 18 + lm + kx) * 192;
#pragma unroll
    for (int c6 = 0; c6 < 6; ++c6, ++kc) {
      const int ci0 = c6 * 32;
      v16h fa;
#pragma unroll
      for (int h = 0; h < 16; ++h) {
        int hk = h + 16 * g;
        fa[h] = prow[ci0 + hk];
      }
      acc0 = wmma_f16(fa, frag_b_packed(Wk + (kc * 4 + nt0) * 512), acc0);
      acc1 = wmma_f16(fa, frag_b_packed(Wk + (kc * 4 + nt0 + 1) * 512), acc1);
    }
  }
  {
    const int nb0 = nt0 * 16;
    float b0 = bias[nb0 + lm], b1 = bias[nb0 + 16 + lm];
#pragma unroll
    for (int r = 0; r < 8; ++r) {
      int pix = mi * 16 + r + 8 * g;
      float v0 = acc0[r] + b0;
      float v1 = acc1[r] + b1;
      if (doLrelu) { v0 = LREL(v0); v1 = LREL(v1); }
      Of[pix * 64 + nb0 + lm] = v0;
      Of[pix * 64 + nb0 + 16 + lm] = v1;
    }
  }
  __syncthreads();
  for (int e = tid; e < 4096; e += 256) {
    int c = e >> 6, m = e & 63, rr = m >> 4, cc = m & 15;
    outp[((size_t)(b * 64 + c) * H + (h0 + rr)) * W + (w0 + cc)] = Of[m * 64 + c];
  }
}

// ---------------------------------------------------------------------------
// leaky-relu then bilinear x2 upsample (half-pixel centers, edge clamp)
// ---------------------------------------------------------------------------
__global__ void lrelu_up2_kernel(const float* __restrict__ in, float* __restrict__ out,
                                 int Hin, int Win, int total) {
  int e = blockIdx.x * 256 + threadIdx.x;
  if (e >= total) return;
  int Wo = Win * 2, Ho = Hin * 2;
  int ow = e % Wo; int t = e / Wo; int oh = t % Ho; int bc = t / Ho;
  float fy = oh * 0.5f - 0.25f, fx = ow * 0.5f - 0.25f;
  int y0 = (int)floorf(fy), x0 = (int)floorf(fx);
  float wy = fy - (float)y0, wx2 = fx - (float)x0;
  int y0c = max(y0, 0), y1c = min(y0 + 1, Hin - 1);
  int x0c = max(x0, 0), x1c = min(x0 + 1, Win - 1);
  const float* p = in + (size_t)bc * Hin * Win;
  float v00 = LREL(p[y0c * Win + x0c]);
  float v01 = LREL(p[y0c * Win + x1c]);
  float v10 = LREL(p[y1c * Win + x0c]);
  float v11 = LREL(p[y1c * Win + x1c]);
  out[e] = (1.0f - wy) * ((1.0f - wx2) * v00 + wx2 * v01) +
           wy * ((1.0f - wx2) * v10 + wx2 * v11);
}

// ---------------------------------------------------------------------------
// weight repack kernels (f32 -> fragment-ready f16)
// qkvp layout per matrix: [j(3)][kc(2)][ni(4)][lane(32)][h(16)]  (4096/level)
// ---------------------------------------------------------------------------
__global__ void repack_qkvp_kernel(const float* __restrict__ qw, const float* __restrict__ kw,
                                   const float* __restrict__ vw, const float* __restrict__ pw,
                                   _Float16* __restrict__ dst) {
  int e = blockIdx.x * 256 + threadIdx.x;
  if (e >= 4 * 3 * 4096) return;
  int mat = e / 12288, r = e % 12288;
  int j = r / 4096, r2 = r % 4096;
  int kc = r2 >> 11;
  int ni = (r2 >> 9) & 3;
  int lane = (r2 >> 4) & 31;
  int h = r2 & 15;
  int k = kc * 32 + 16 * (lane >> 4) + h;  // input channel
  int n = ni * 16 + (lane & 15);           // output channel
  const float* src = (mat == 0) ? qw : (mat == 1) ? kw : (mat == 2) ? vw : pw;
  dst[e] = (_Float16)src[j * 4096 + n * 64 + k];
}

// fc layout per level: [kc(54)][ni(4)][lane(32)][h(16)], k = tap*192 + ci
__global__ void repack_fc_kernel(const float* __restrict__ fcw, _Float16* __restrict__ dst) {
  int e = blockIdx.x * 256 + threadIdx.x;
  if (e >= 2 * 1728 * 64) return;
  int j = e / 110592, r = e % 110592;
  int kc = r / 2048, r2 = r % 2048;
  int ni = r2 >> 9;
  int lane = (r2 >> 4) & 31;
  int h = r2 & 15;
  int k = kc * 32 + 16 * (lane >> 4) + h;
  int tap = k / 192, ci = k % 192;
  int ky = tap / 3, kx = tap % 3;
  int o = ni * 16 + (lane & 15);
  dst[e] = (_Float16)fcw[(((size_t)(j * 64 + o) * 192 + ci) * 3 + ky) * 3 + kx];
}

// ---------------------------------------------------------------------------
extern "C" void kernel_launch(void* const* d_in, const int* in_sizes, int n_in,
                              void* d_out, int out_size, void* d_ws, size_t ws_size,
                              hipStream_t stream) {
  const float* ref1 = (const float*)d_in[0];
  const float* ref2 = (const float*)d_in[1];
  const float* ref3 = (const float*)d_in[2];
  const float* ta1  = (const float*)d_in[3];
  const float* ta2  = (const float*)d_in[4];
  const float* ta3  = (const float*)d_in[5];
  const float* q_w  = (const float*)d_in[6];
  const float* q_b  = (const float*)d_in[7];
  const float* k_w  = (const float*)d_in[8];
  const float* k_b  = (const float*)d_in[9];
  const float* v_w  = (const float*)d_in[10];
  const float* v_b  = (const float*)d_in[11];
  const float* p_w  = (const float*)d_in[12];
  const float* p_b  = (const float*)d_in[13];
  const float* btab = (const float*)d_in[14];
  const float* fc_w = (const float*)d_in[15];
  const float* fc_b = (const float*)d_in[16];

  char* ws = (char*)d_ws;
  _Float16* Wqkvp = (_Float16*)ws;                 // 49152 halves
  _Float16* Wfc   = (_Float16*)(ws + (1 << 20));   // 221184 halves
  float* fbase = (float*)(ws + (2u << 20));
  size_t off = 0;
  float* attn3  = fbase + off; off += (size_t)256 * 4096;
  float* attn2  = fbase + off; off += (size_t)1024 * 4096;
  float* feat3  = fbase + off; off += (size_t)4 * 64 * 64 * 64;
  float* up2    = fbase + off; off += (size_t)4 * 64 * 128 * 128;
  float* feat2a = fbase + off; off += (size_t)4 * 64 * 128 * 128;
  float* atr2   = fbase + off; off += (size_t)4 * 64 * 128 * 128;
  float* feat2  = fbase + off; off += (size_t)4 * 64 * 128 * 128;
  float* up1    = fbase + off; off += (size_t)4 * 64 * 256 * 256;
  float* feat1a = fbase + off; off += (size_t)4 * 64 * 256 * 256;
  float* atr1   = fbase + off; off += (size_t)4 * 64 * 256 * 256;
  float* outF = (float*)d_out;

  repack_qkvp_kernel<<<(49152 + 255) / 256, 256, 0, stream>>>(q_w, k_w, v_w, p_w, Wqkvp);
  repack_fc_kernel<<<(221184 + 255) / 256, 256, 0, stream>>>(fc_w, Wfc);

  const _Float16* Wq = Wqkvp;
  const _Float16* Wk = Wqkvp + 12288;
  const _Float16* Wv = Wqkvp + 24576;
  const _Float16* Wp = Wqkvp + 36864;

  // ---- level 3 (j=2, 64x64) ----
  {
    int j = 2, H = 64, W = 64;
    win_attn_kernel<<<4 * (H / 8) * (W / 8), 256, 0, stream>>>(
        ref3, ta3, Wq + j * 4096, Wk + j * 4096, Wv + j * 4096, Wp + j * 4096,
        q_b + j * 64, k_b + j * 64, v_b + j * 64, p_b + j * 64, btab + j * 900,
        feat3, attn3, H, W);
    int total = 4 * 64 * (2 * H) * (2 * W);
    lrelu_up2_kernel<<<(total + 255) / 256, 256, 0, stream>>>(feat3, up2, H, W, total);
  }
  // ---- level 2 (j=1, 128x128) ----
  {
    int j = 1, H = 128, W = 128;
    win_attn_kernel<<<4 * (H / 8) * (W / 8), 256, 0, stream>>>(
        ref2, ta2, Wq + j * 4096, Wk + j * 4096, Wv + j * 4096, Wp + j * 4096,
        q_b + j * 64, k_b + j * 64, v_b + j * 64, p_b + j * 64, btab + j * 900,
        feat2a, attn2, H, W);
    atttrans_kernel<<<4 * (H / 16) * (W / 16), 256, 0, stream>>>(ta2, attn3, atr2, H, W);
    conv3_kernel<<<4 * (H / 4) * (W / 16), 256, 0, stream>>>(
        feat2a, up2, atr2, Wfc + (size_t)j * 110592, fc_b + j * 64, feat2, H, W, 0);
    int total = 4 * 64 * (2 * H) * (2 * W);
    lrelu_up2_kernel<<<(total + 255) / 256, 256, 0, stream>>>(feat2, up1, H, W, total);
  }
  // ---- level 1 (j=0, 256x256) ----
  {
    int H = 256, W = 256;
    win_attn_kernel<<<4 * (H / 8) * (W / 8), 256, 0, stream>>>(
        ref1, ta1, Wq, Wk, Wv, Wp, q_b, k_b, v_b, p_b, btab,
        feat1a, nullptr, H, W);
    atttrans_kernel<<<4 * (H / 16) * (W / 16), 256, 0, stream>>>(ta1, attn2, atr1, H, W);
    conv3_kernel<<<4 * (H / 4) * (W / 16), 256, 0, stream>>>(
        feat1a, up1, atr1, Wfc, fc_b, outF, H, W, 1);
  }
  (void)in_sizes; (void)n_in; (void)out_size; (void)ws_size;
}
